// DMPNN_Encoder_5119601016899
// MI455X (gfx1250) — matrix operand order repaired and checked by reading, not compiled, not run on hardware
//
#include <hip/hip_runtime.h>
#include <hip/hip_bf16.h>

typedef __attribute__((ext_vector_type(16))) __bf16 v16bf;
typedef __attribute__((ext_vector_type(8)))  float  v8f;

union Frag { v16bf v; unsigned int u[8]; uint4 q[2]; };   // natural align = 32

__device__ inline unsigned short f2bf(float f) {
    union { float f; unsigned int u; } c; c.f = f;
    unsigned int u = c.u;
    u += 0x7fffu + ((u >> 16) & 1u);      // round-to-nearest-even
    return (unsigned short)(u >> 16);
}

__device__ inline void atomAddF(float* p, float v) {
    (void)__hip_atomic_fetch_add(p, v, __ATOMIC_RELAXED, __HIP_MEMORY_SCOPE_AGENT);
}

// ---------------------------------------------------------------------------
// Weight convert+transpose: W [K x 256] f32  ->  Wt [256 x Kpad] bf16 (zero pad)
// Row-major N x Kpad means each lane's B fragment (K = hi*8 + 0..7, 16 + hi*8 + 0..7) is two
// 16-byte chunks -> loaded straight from global as 2x b128.
// ---------------------------------------------------------------------------
__global__ __launch_bounds__(256)
void convert_wt_kernel(const float* __restrict__ W, int K, int Kpad,
                       unsigned short* __restrict__ Wt) {
    int idx = blockIdx.x * 256 + threadIdx.x;
    int total = 256 * Kpad;
    if (idx >= total) return;
    int n = idx / Kpad;
    int k = idx - n * Kpad;
    float v = (k < K) ? W[(size_t)k * 256 + n] : 0.0f;
    Wt[idx] = f2bf(v);
}

// ---------------------------------------------------------------------------
// WMMA GEMM:  C[M x 256] = act( concat(A0[gidx? gidx[m]:m], A1[m]) @ W + bias ) (+res)
// A staged in LDS in *fragment order*: chunk c = (kblk*2 + rowTile)*32 + lane,
// each chunk = 16 bf16 = one lane's A fragment for that k-block.
// Compute loop: A = 2x ds_load_b128 (linear stride), B = 2x global_load_b128.
// Block: 256 threads = 8 waves; tile 32 rows x 64 cols; wave = one 16x16 D tile.
// ---------------------------------------------------------------------------
__global__ __launch_bounds__(256)
void wmma_gemm_kernel(const float* __restrict__ A0, int K0, const int* __restrict__ gidx,
                      const float* __restrict__ A1, int K1,
                      int M, int Kpad,
                      const unsigned short* __restrict__ Wt,
                      const float* __restrict__ bias,
                      const float* __restrict__ residual,
                      int do_relu,
                      float* __restrict__ C) {
    extern __shared__ unsigned short sA[];   // 2*Kpad chunks x 16 bf16 = 64*Kpad bytes

    const int tid     = threadIdx.x;
    const int rowBase = blockIdx.x * 32;
    const int colBase = blockIdx.y * 64;
    const int Ktot    = K0 + K1;
    const int nchunks = 2 * Kpad;            // (Kpad/32) kblks * 2 rowTiles * 32 lanes

    // ---- stage A tile in fragment order (gather + concat + f32->bf16) ----
    for (int c = tid; c < nchunks; c += 256) {
        int lc   = c & 31;                   // lane within fragment tile
        int rt   = (c >> 5) & 1;             // row tile (0..1)
        int kblk = c >> 6;                   // k block (32 wide)
        int m    = rowBase + rt * 16 + (lc & 15);
        int hic  = lc >> 4;
        int srow = -1;
        if (m < M) srow = gidx ? gidx[m] : m;
        unsigned int vals[8];
#pragma unroll
        for (int p = 0; p < 8; ++p) {
            // A 16x32 bf16 layout: VGPR p holds K = (p>=4)*16 + hi*8 + 2*(p&3) (+1)
            int k0i = kblk * 32 + ((p >> 2) << 4) + (hic << 3) + ((p & 3) << 1);
            float f0 = 0.0f, f1 = 0.0f;
            if (srow >= 0) {
                if (k0i < K0)        f0 = A0[(size_t)srow * K0 + k0i];
                else if (k0i < Ktot) f0 = A1[(size_t)m * K1 + (k0i - K0)];
                int k1i = k0i + 1;
                if (k1i < K0)        f1 = A0[(size_t)srow * K0 + k1i];
                else if (k1i < Ktot) f1 = A1[(size_t)m * K1 + (k1i - K0)];
            }
            vals[p] = (unsigned int)f2bf(f0) | ((unsigned int)f2bf(f1) << 16);
        }
        uint4* d = (uint4*)(sA + (size_t)c * 16);
        d[0] = make_uint4(vals[0], vals[1], vals[2], vals[3]);
        d[1] = make_uint4(vals[4], vals[5], vals[6], vals[7]);
    }
    __syncthreads();

    const int wave = tid >> 5, lane = tid & 31;
    const int wr = wave >> 2, wc = wave & 3;           // 2 row-tiles x 4 col-tiles
    const int l16 = lane & 15, hi = lane >> 4;

    const uint4* aP   = (const uint4*)sA + ((size_t)wr * 32 + lane) * 2;
    const uint4* bRow = (const uint4*)(Wt + (size_t)(colBase + wc * 16 + l16) * Kpad);

    v8f acc = {};
    const int nk = Kpad >> 5;
    for (int kb = 0; kb < nk; ++kb) {
        Frag a, b;
        a.q[0] = aP[0];
        a.q[1] = aP[1];
        b.q[0] = bRow[kb * 4 + hi];
        b.q[1] = bRow[kb * 4 + 2 + hi];
        acc = __builtin_amdgcn_wmma_f32_16x16x32_bf16(false, a.v, false, b.v,
                                                      (short)0, acc, false, false);
        aP += 128;                                     // 64 chunks = 128 uint4 per kblk
    }

    // ---- epilogue: D layout VGPR r -> M = hi*8 + r, N = l16 ----
    const int outCol = colBase + wc * 16 + l16;
    float bv = bias ? bias[outCol] : 0.0f;
#pragma unroll
    for (int r = 0; r < 8; ++r) {
        int m = rowBase + wr * 16 + hi * 8 + r;
        if (m < M) {
            float v = acc[r] + bv;
            if (do_relu) v = fmaxf(v, 0.0f);
            if (residual) v += residual[(size_t)m * 256 + outCol];
            C[(size_t)m * 256 + outCol] = v;
        }
    }
}

// ---------------------------------------------------------------------------
// Attention logits: one wave per triplet.
// att = exp(leaky_relu(sum_d q[kj]*k[ji], 0.2)) per head; accumulate denom[ji].
// ---------------------------------------------------------------------------
__global__ __launch_bounds__(256)
void att_logits_kernel(const float* __restrict__ q, const float* __restrict__ kf,
                       const int* __restrict__ idx_kj, const int* __restrict__ idx_ji,
                       float* __restrict__ att_raw, float* __restrict__ denom, int T) {
    int t = (blockIdx.x * 256 + threadIdx.x) >> 5;
    int lane = threadIdx.x & 31;
    if (t >= T) return;
    int kj = idx_kj[t], ji = idx_ji[t];
    const float4* qr = (const float4*)(q  + (size_t)kj * 256);
    const float4* kr = (const float4*)(kf + (size_t)ji * 256);
    float4 q0 = qr[lane * 2], q1 = qr[lane * 2 + 1];
    float4 k0 = kr[lane * 2], k1 = kr[lane * 2 + 1];
    float s = q0.x * k0.x + q0.y * k0.y + q0.z * k0.z + q0.w * k0.w
            + q1.x * k1.x + q1.y * k1.y + q1.z * k1.z + q1.w * k1.w;
    // reduce within 4-lane group (one head = 32 channels = 4 lanes x 8)
    s += __shfl_xor(s, 1);
    s += __shfl_xor(s, 2);
    if ((lane & 3) == 0) {
        int h = lane >> 2;
        float a = (s > 0.0f) ? s : 0.2f * s;
        float e = __expf(a);
        att_raw[(size_t)t * 8 + h] = e;
        atomAddF(&denom[(size_t)ji * 8 + h], e);
    }
}

// ---------------------------------------------------------------------------
// Normalize + message scatter: v_clone[ji] += v[kj] * att/denom. One wave/triplet.
// ---------------------------------------------------------------------------
__global__ __launch_bounds__(256)
void scatter_v_kernel(const float* __restrict__ v, const int* __restrict__ idx_kj,
                      const int* __restrict__ idx_ji, const float* __restrict__ att_raw,
                      const float* __restrict__ denom, float* __restrict__ v_clone, int T) {
    int t = (blockIdx.x * 256 + threadIdx.x) >> 5;
    int lane = threadIdx.x & 31;
    if (t >= T) return;
    int kj = idx_kj[t], ji = idx_ji[t];
    int h = lane >> 2;
    float scale = att_raw[(size_t)t * 8 + h] / denom[(size_t)ji * 8 + h];
    const float4* vr = (const float4*)(v + (size_t)kj * 256);
    float4 v0 = vr[lane * 2], v1 = vr[lane * 2 + 1];
    float* out = v_clone + (size_t)ji * 256 + lane * 8;
    atomAddF(out + 0, v0.x * scale); atomAddF(out + 1, v0.y * scale);
    atomAddF(out + 2, v0.z * scale); atomAddF(out + 3, v0.w * scale);
    atomAddF(out + 4, v1.x * scale); atomAddF(out + 5, v1.y * scale);
    atomAddF(out + 6, v1.z * scale); atomAddF(out + 7, v1.w * scale);
}

// ---------------------------------------------------------------------------
// Edge -> node segment sum: fsum[dst[e]] += feats[e]. One wave per edge.
// ---------------------------------------------------------------------------
__global__ __launch_bounds__(256)
void edge_sum_kernel(const float* __restrict__ feats, const int* __restrict__ dst,
                     float* __restrict__ fsum, int E) {
    int e = (blockIdx.x * 256 + threadIdx.x) >> 5;
    int lane = threadIdx.x & 31;
    if (e >= E) return;
    int d = dst[e];
    const float4* fr = (const float4*)(feats + (size_t)e * 256);
    float4 f0 = fr[lane * 2], f1 = fr[lane * 2 + 1];
    float* out = fsum + (size_t)d * 256 + lane * 8;
    atomAddF(out + 0, f0.x); atomAddF(out + 1, f0.y);
    atomAddF(out + 2, f0.z); atomAddF(out + 3, f0.w);
    atomAddF(out + 4, f1.x); atomAddF(out + 5, f1.y);
    atomAddF(out + 6, f1.z); atomAddF(out + 7, f1.w);
}

// ---------------------------------------------------------------------------
extern "C" void kernel_launch(void* const* d_in, const int* in_sizes, int n_in,
                              void* d_out, int out_size, void* d_ws, size_t ws_size,
                              hipStream_t stream) {
    const float* atom = (const float*)d_in[0];
    const float* edge = (const float*)d_in[1];
    const float* W_i  = (const float*)d_in[2];
    const float* Wq   = (const float*)d_in[3];
    const float* Wk   = (const float*)d_in[4];
    const float* Wv   = (const float*)d_in[5];
    const float* L1w  = (const float*)d_in[6];
    const float* L1b  = (const float*)d_in[7];
    const float* L2w  = (const float*)d_in[8];
    const float* L2b  = (const float*)d_in[9];
    const float* Wo   = (const float*)d_in[10];
    const float* bo   = (const float*)d_in[11];
    const int* src    = (const int*)d_in[12];
    const int* dst    = (const int*)d_in[13];
    const int* idx_kj = (const int*)d_in[14];
    const int* idx_ji = (const int*)d_in[15];

    const int Nn = in_sizes[0] / 133;   // nodes
    const int E  = in_sizes[1] / 14;    // edges
    const int T  = in_sizes[14];        // triplets
    const int L  = 2;                   // RADIUS - 1

    // ---- workspace layout ----
    char* w = (char*)d_ws;
    auto take = [&](size_t bytes) -> char* {
        char* p = w; w += (bytes + 255) & ~(size_t)255; return p;
    };
    float* feats   = (float*)take((size_t)E * 256 * 4);
    float* qb      = (float*)take((size_t)E * 256 * 4);
    float* kb      = (float*)take((size_t)E * 256 * 4);
    float* vb      = (float*)take((size_t)E * 256 * 4);
    float* vclone  = (float*)take((size_t)E * 256 * 4);
    float* tmp     = (float*)take((size_t)E * 256 * 4);
    float* att_raw = (float*)take((size_t)T * 8 * 4);
    float* denom   = (float*)take((size_t)E * 8 * 4);
    float* fsum    = (float*)take((size_t)Nn * 256 * 4);
    unsigned short* wtbuf = (unsigned short*)take((size_t)256 * 416 * 2);

    auto conv = [&](const float* Wsrc, int K, int Kpad) {
        int total = 256 * Kpad;
        convert_wt_kernel<<<(total + 255) / 256, 256, 0, stream>>>(Wsrc, K, Kpad, wtbuf);
    };
    auto gemm = [&](const float* A0, int K0, const int* gi, const float* A1, int K1,
                    int M, int Kpad, const float* bias, const float* resid,
                    int relu, float* C) {
        dim3 grid((M + 31) / 32, 4);
        size_t lds = (size_t)32 * Kpad * 2;   // A tile only (fragment-ordered)
        wmma_gemm_kernel<<<grid, 256, lds, stream>>>(A0, K0, gi, A1, K1, M, Kpad,
                                                     wtbuf, bias, resid, relu, C);
    };

    // ---- init: feats = relu(concat(atom[src], edge) @ W_i) ----
    conv(W_i, 147, 160);
    gemm(atom, 133, src, edge, 14, E, 160, nullptr, nullptr, 1, feats);

    const int tripBlocks = (T + 7) / 8;   // 8 waves per block, one triplet per wave
    const int edgeBlocks = (E + 7) / 8;

    for (int l = 0; l < L; ++l) {
        const size_t wo = (size_t)l * 256 * 256;
        // q,k,v projections (no bias, no act)
        conv(Wq + wo, 256, 256); gemm(feats, 256, nullptr, nullptr, 0, E, 256, nullptr, nullptr, 0, qb);
        conv(Wk + wo, 256, 256); gemm(feats, 256, nullptr, nullptr, 0, E, 256, nullptr, nullptr, 0, kb);
        conv(Wv + wo, 256, 256); gemm(feats, 256, nullptr, nullptr, 0, E, 256, nullptr, nullptr, 0, vb);

        // attention logits + edge-softmax denominator
        (void)hipMemsetAsync(denom, 0, (size_t)E * 8 * 4, stream);
        att_logits_kernel<<<tripBlocks, 256, 0, stream>>>(qb, kb, idx_kj, idx_ji,
                                                          att_raw, denom, T);
        // message aggregation
        (void)hipMemsetAsync(vclone, 0, (size_t)E * 256 * 4, stream);
        scatter_v_kernel<<<tripBlocks, 256, 0, stream>>>(vb, idx_kj, idx_ji,
                                                         att_raw, denom, vclone, T);
        // residual MLP: feats = v + relu(relu(vclone@L1+b1)@L2+b2)
        conv(L1w + wo, 256, 256);
        gemm(vclone, 256, nullptr, nullptr, 0, E, 256, L1b + (size_t)l * 256, nullptr, 1, tmp);
        conv(L2w + wo, 256, 256);
        gemm(tmp, 256, nullptr, nullptr, 0, E, 256, L2b + (size_t)l * 256, vb, 1, feats);
    }

    // ---- edge -> node sum, then out = relu(concat(atom, fsum) @ Wo + bo) ----
    (void)hipMemsetAsync(fsum, 0, (size_t)Nn * 256 * 4, stream);
    edge_sum_kernel<<<edgeBlocks, 256, 0, stream>>>(feats, dst, fsum, E);
    conv(Wo, 389, 416);
    gemm(atom, 133, nullptr, fsum, 256, Nn, 416, bo, nullptr, 1, (float*)d_out);
}
